// MLCDSdpaAttention_3822520893530
// MI455X (gfx1250) — compile-verified, matrix operand order checked by reading, not run on hardware
//
#include <hip/hip_runtime.h>

typedef __bf16 bf16;
typedef __attribute__((ext_vector_type(2)))  __bf16 v2bf;
typedef __attribute__((ext_vector_type(4)))  __bf16 v4bf;
typedef __attribute__((ext_vector_type(8)))  __bf16 v8bf;
typedef __attribute__((ext_vector_type(16))) __bf16 v16bf;
typedef __attribute__((ext_vector_type(8)))  float  v8f;

constexpr int B = 8, S = 1024, E = 1024, H = 16, D = 64, HALF = 32;
constexpr int BS = B * S;

#define WMMA_BF16(a, b, c) \
  __builtin_amdgcn_wmma_f32_16x16x32_bf16(false, (a), false, (b), (short)0, (c), false, false)

__device__ __forceinline__ v16bf gather16(const bf16* p, int o0, int o1) {
  v8bf lo = *(const v8bf*)(p + o0);
  v8bf hi = *(const v8bf*)(p + o1);
  v16bf r;
#pragma unroll
  for (int i = 0; i < 8; ++i) { r[i] = lo[i]; r[i + 8] = hi[i]; }
  return r;
}

__device__ __forceinline__ float redmax16(float v) {
#pragma unroll
  for (int m = 1; m < 16; m <<= 1) v = fmaxf(v, __shfl_xor(v, m, 32));
  return v;
}
__device__ __forceinline__ float redsum16(float v) {
#pragma unroll
  for (int m = 1; m < 16; m <<= 1) v += __shfl_xor(v, m, 32);
  return v;
}

// ---------------------------------------------------------------------------
// Kernel 1: fused QKV projection + bias + RoPE, output bf16 in [b,h,s,d]
// grid: (BS/128, H, 3)   block: 128 (4 waves, 32 rows each)
// ---------------------------------------------------------------------------
__global__ __launch_bounds__(128) void qkv_rope_kernel(
    const float* __restrict__ hs, const float* __restrict__ rope,
    const float* __restrict__ wq, const float* __restrict__ bq,
    const float* __restrict__ wk, const float* __restrict__ bk,
    const float* __restrict__ wv, const float* __restrict__ bv,
    bf16* __restrict__ outQ, bf16* __restrict__ outK, bf16* __restrict__ outV) {
  constexpr int ASTR = 48, BSTR = 48;                // 96B row stride: 16B aligned
  __shared__ __align__(16) bf16 As[128 * ASTR];      // 128 rows x 32 k
  __shared__ __align__(16) bf16 Bt[64 * BSTR];       // 64 n  x 32 k (transposed)

  const int z = blockIdx.z;
  const float* w   = (z == 0) ? wq : (z == 1) ? wk : wv;
  const float* bia = (z == 0) ? bq : (z == 1) ? bk : bv;
  bf16* dst        = (z == 0) ? outQ : (z == 1) ? outK : outV;

  const int rowBase = blockIdx.x * 128;
  const int h       = blockIdx.y;                    // 64-col tile == one head
  const int colBase = h * 64;

  const int tid  = threadIdx.x;
  const int wv32 = tid >> 5;
  const int lane = tid & 31;
  const int l15  = lane & 15;
  const int lh   = (lane >> 4) & 1;
  const int o0   = lh ? 8 : 0;                       // bf16 A/B frag K-chunk offsets
  const int o1   = 16 + o0;

  v8f acc[2][4] = {};

  for (int kb = 0; kb < E; kb += 32) {
    if (kb + 32 < E) {                               // prefetch next k-tile
      __builtin_prefetch(hs + (size_t)(rowBase + tid) * E + kb + 32, 0, 1);
      __builtin_prefetch(w + (size_t)(kb + 32 + (tid & 31)) * E + colBase, 0, 1);
    }
    __syncthreads();
#pragma unroll
    for (int it = 0; it < 8; ++it) {                 // A: 128x32 f32 -> bf16 packed
      int idx = tid + it * 128;
      int r = idx >> 3, c = (idx & 7) << 2;
      float4 f = *(const float4*)(hs + (size_t)(rowBase + r) * E + kb + c);
      v4bf t; t[0] = (bf16)f.x; t[1] = (bf16)f.y; t[2] = (bf16)f.z; t[3] = (bf16)f.w;
      *(v4bf*)&As[r * ASTR + c] = t;                 // one ds_store_b64
    }
#pragma unroll
    for (int it = 0; it < 2; ++it) {                 // B: 32x64 -> bf16 transposed, pair-packed
      int idx = tid + it * 128;
      int kk = (idx >> 4) << 1, n = (idx & 15) << 2;
      float4 f0 = *(const float4*)(w + (size_t)(kb + kk) * E + colBase + n);
      float4 f1 = *(const float4*)(w + (size_t)(kb + kk + 1) * E + colBase + n);
      v2bf p;
      p[0] = (bf16)f0.x; p[1] = (bf16)f1.x; *(v2bf*)&Bt[(n + 0) * BSTR + kk] = p;
      p[0] = (bf16)f0.y; p[1] = (bf16)f1.y; *(v2bf*)&Bt[(n + 1) * BSTR + kk] = p;
      p[0] = (bf16)f0.z; p[1] = (bf16)f1.z; *(v2bf*)&Bt[(n + 2) * BSTR + kk] = p;
      p[0] = (bf16)f0.w; p[1] = (bf16)f1.w; *(v2bf*)&Bt[(n + 3) * BSTR + kk] = p;
    }
    __syncthreads();

    v16bf af[2];
    af[0] = gather16(&As[(wv32 * 32 + l15) * ASTR], o0, o1);
    af[1] = gather16(&As[(wv32 * 32 + 16 + l15) * ASTR], o0, o1);
    v16bf bfr[4];
#pragma unroll
    for (int nf = 0; nf < 4; ++nf)
      bfr[nf] = gather16(&Bt[(nf * 16 + l15) * BSTR], o0, o1);
#pragma unroll
    for (int ms = 0; ms < 2; ++ms)
#pragma unroll
      for (int nf = 0; nf < 4; ++nf)
        acc[ms][nf] = WMMA_BF16(af[ms], bfr[nf], acc[ms][nf]);
  }

  float bi[4];
#pragma unroll
  for (int nf = 0; nf < 4; ++nf) bi[nf] = bia[colBase + nf * 16 + l15];

#pragma unroll
  for (int ms = 0; ms < 2; ++ms) {
#pragma unroll
    for (int i = 0; i < 8; ++i) {
      int m = ms * 16 + i + lh * 8;
      int g = rowBase + wv32 * 32 + m;               // global row = b*S + s
      int s = g & (S - 1);
      int bb = g >> 10;
      float a0 = acc[ms][0][i] + bi[0];
      float a1 = acc[ms][1][i] + bi[1];
      float a2 = acc[ms][2][i] + bi[2];
      float a3 = acc[ms][3][i] + bi[3];
      if (z < 2) {                                   // RoPE for Q and K
        float fl = rope[s * HALF + l15];
        float fh = rope[s * HALF + 16 + l15];
        float cl = __cosf(fl), sl = __sinf(fl);
        float ch = __cosf(fh), sh = __sinf(fh);
        float r0 = a0 * cl - a2 * sl;                // pair (d, d+32) = frags (nf, nf^2)
        float r1 = a1 * ch - a3 * sh;
        float r2 = a2 * cl + a0 * sl;
        float r3 = a3 * ch + a1 * sh;
        a0 = r0; a1 = r1; a2 = r2; a3 = r3;
      }
      bf16* o = dst + (((size_t)bb * H + h) * S + s) * D;
      o[0 * 16 + l15] = (bf16)a0;
      o[1 * 16 + l15] = (bf16)a1;
      o[2 * 16 + l15] = (bf16)a2;
      o[3 * 16 + l15] = (bf16)a3;
    }
  }
}

// ---------------------------------------------------------------------------
// Kernel 2: flash attention per (b,h); 128 q-rows per block, 64-wide k tiles
// grid: (S/128, B*H)   block: 256 (8 waves, 16 q-rows each)
// ---------------------------------------------------------------------------
__global__ __launch_bounds__(256) void flash_attn_kernel(
    const bf16* __restrict__ Q, const bf16* __restrict__ K,
    const bf16* __restrict__ V, bf16* __restrict__ O) {
  constexpr int KSTR = 72, VSTR = 72, PSTR = 72;     // 144B strides: 16B aligned
  __shared__ __align__(16) bf16 Ks[64 * KSTR];       // [kpos][d]
  __shared__ __align__(16) bf16 Vt[64 * VSTR];       // [d][kpos]
  __shared__ __align__(16) bf16 Ps[8 * 16 * PSTR];   // per-wave P tile (16x64)

  const int bh = blockIdx.y;
  const int bb = bh >> 4, h = bh & (H - 1);
  const int qbase = blockIdx.x * 128;

  const int tid = threadIdx.x, wv32 = tid >> 5, lane = tid & 31;
  const int l15 = lane & 15, lh = (lane >> 4) & 1;
  const int o0 = lh ? 8 : 0, o1 = 16 + o0;

  const bf16* qp = Q + ((size_t)bh * S + qbase + wv32 * 16 + l15) * D;
  v16bf qf[2];
  qf[0] = gather16(qp, o0, o1);                      // Q A-frags, d 0..31 / 32..63
  qf[1] = gather16(qp, 32 + o0, 32 + o1);

  const bf16* kp = K + (size_t)bh * S * D;
  const bf16* vp = V + (size_t)bh * S * D;

  float maxv[8], sumv[8];
#pragma unroll
  for (int i = 0; i < 8; ++i) { maxv[i] = -1e30f; sumv[i] = 0.f; }
  v8f acc[4] = {};
  const int pbase = wv32 * 16 * PSTR;

  for (int kt = 0; kt < S; kt += 64) {
    if (kt + 64 < S) {                               // prefetch next K/V rows
      __builtin_prefetch(kp + (size_t)(kt + 64 + (tid & 63)) * D, 0, 1);
      __builtin_prefetch(vp + (size_t)(kt + 64 + (tid & 63)) * D, 0, 1);
    }
    __syncthreads();
#pragma unroll
    for (int it = 0; it < 2; ++it) {                 // K tile: 64x64 bulk copy
      int idx = tid + it * 256;
      int r = idx >> 3, c = (idx & 7) << 3;
      v8bf kv = *(const v8bf*)(kp + (size_t)(kt + r) * D + c);
      *(v8bf*)&Ks[r * KSTR + c] = kv;
    }
    {                                                // V transposed, row-pair packed
      int r = (tid >> 3) << 1, c = (tid & 7) << 3;
      v8bf v0 = *(const v8bf*)(vp + (size_t)(kt + r) * D + c);
      v8bf v1 = *(const v8bf*)(vp + (size_t)(kt + r + 1) * D + c);
#pragma unroll
      for (int j = 0; j < 8; ++j) {
        v2bf p; p[0] = v0[j]; p[1] = v1[j];
        *(v2bf*)&Vt[(c + j) * VSTR + r] = p;         // one ds_store_b32
      }
    }
    __syncthreads();

    v8f sf[4] = {};                                  // scores 16x64 (4 C frags)
#pragma unroll
    for (int ks = 0; ks < 2; ++ks) {
      v16bf bk[4];
#pragma unroll
      for (int nf = 0; nf < 4; ++nf)
        bk[nf] = gather16(&Ks[(nf * 16 + l15) * KSTR], ks * 32 + o0, ks * 32 + o1);
#pragma unroll
      for (int nf = 0; nf < 4; ++nf)
        sf[nf] = WMMA_BF16(qf[ks], bk[nf], sf[nf]);
    }

#pragma unroll
    for (int i = 0; i < 8; ++i) {                    // online softmax per row
      float s0 = sf[0][i] * 0.125f, s1 = sf[1][i] * 0.125f;   // 1/sqrt(64)
      float s2 = sf[2][i] * 0.125f, s3 = sf[3][i] * 0.125f;
      float m = redmax16(fmaxf(fmaxf(s0, s1), fmaxf(s2, s3)));
      float newm = fmaxf(maxv[i], m);
      float alpha = __expf(maxv[i] - newm);
      maxv[i] = newm;
      float p0 = __expf(s0 - newm), p1 = __expf(s1 - newm);
      float p2 = __expf(s2 - newm), p3 = __expf(s3 - newm);
      sumv[i] = sumv[i] * alpha + redsum16((p0 + p1) + (p2 + p3));
      acc[0][i] *= alpha; acc[1][i] *= alpha;
      acc[2][i] *= alpha; acc[3][i] *= alpha;
      int mr = i + lh * 8;                           // C layout -> LDS [row][col]
      Ps[pbase + mr * PSTR + 0 * 16 + l15] = (bf16)p0;
      Ps[pbase + mr * PSTR + 1 * 16 + l15] = (bf16)p1;
      Ps[pbase + mr * PSTR + 2 * 16 + l15] = (bf16)p2;
      Ps[pbase + mr * PSTR + 3 * 16 + l15] = (bf16)p3;
    }
    __syncthreads();

    v16bf pf[2];                                     // P as A-frags (k = 0..31 / 32..63)
    pf[0] = gather16(&Ps[pbase + l15 * PSTR], o0, o1);
    pf[1] = gather16(&Ps[pbase + l15 * PSTR], 32 + o0, 32 + o1);
#pragma unroll
    for (int ks = 0; ks < 2; ++ks) {
      v16bf bv[4];
#pragma unroll
      for (int nf = 0; nf < 4; ++nf)
        bv[nf] = gather16(&Vt[(nf * 16 + l15) * VSTR], ks * 32 + o0, ks * 32 + o1);
#pragma unroll
      for (int nf = 0; nf < 4; ++nf)
        acc[nf] = WMMA_BF16(pf[ks], bv[nf], acc[nf]);
    }
  }

#pragma unroll
  for (int i = 0; i < 8; ++i) {                      // normalize + store [b,s,h*D+d]
    int m = i + lh * 8;
    int srow = qbase + wv32 * 16 + m;
    float inv = 1.0f / sumv[i];
    bf16* o = O + ((size_t)bb * S + srow) * E + h * D;
#pragma unroll
    for (int nf = 0; nf < 4; ++nf)
      o[nf * 16 + l15] = (bf16)(acc[nf][i] * inv);
  }
}

// ---------------------------------------------------------------------------
// Kernel 3: output projection, bf16 A from ws, f32 out + bias
// grid: (BS/128, E/64)   block: 128 (4 waves, 32 rows each)
// ---------------------------------------------------------------------------
__global__ __launch_bounds__(128) void out_proj_kernel(
    const bf16* __restrict__ A, const float* __restrict__ w,
    const float* __restrict__ bia, float* __restrict__ out) {
  constexpr int ASTR = 48, BSTR = 48;
  __shared__ __align__(16) bf16 As[128 * ASTR];
  __shared__ __align__(16) bf16 Bt[64 * BSTR];
  const int rowBase = blockIdx.x * 128;
  const int colBase = blockIdx.y * 64;
  const int tid = threadIdx.x, wv32 = tid >> 5, lane = tid & 31;
  const int l15 = lane & 15, lh = (lane >> 4) & 1;
  const int o0 = lh ? 8 : 0, o1 = 16 + o0;
  v8f acc[2][4] = {};

  for (int kb = 0; kb < E; kb += 32) {
    if (kb + 32 < E) {
      __builtin_prefetch(A + (size_t)(rowBase + tid) * E + kb + 32, 0, 1);
      __builtin_prefetch(w + (size_t)(kb + 32 + (tid & 31)) * E + colBase, 0, 1);
    }
    __syncthreads();
#pragma unroll
    for (int it = 0; it < 4; ++it) {                 // A already bf16: bulk copy
      int idx = tid + it * 128;
      int r = idx >> 2, c = (idx & 3) << 3;
      v8bf av = *(const v8bf*)(A + (size_t)(rowBase + r) * E + kb + c);
      *(v8bf*)&As[r * ASTR + c] = av;
    }
#pragma unroll
    for (int it = 0; it < 2; ++it) {                 // B transposed, pair-packed
      int idx = tid + it * 128;
      int kk = (idx >> 4) << 1, n = (idx & 15) << 2;
      float4 f0 = *(const float4*)(w + (size_t)(kb + kk) * E + colBase + n);
      float4 f1 = *(const float4*)(w + (size_t)(kb + kk + 1) * E + colBase + n);
      v2bf p;
      p[0] = (bf16)f0.x; p[1] = (bf16)f1.x; *(v2bf*)&Bt[(n + 0) * BSTR + kk] = p;
      p[0] = (bf16)f0.y; p[1] = (bf16)f1.y; *(v2bf*)&Bt[(n + 1) * BSTR + kk] = p;
      p[0] = (bf16)f0.z; p[1] = (bf16)f1.z; *(v2bf*)&Bt[(n + 2) * BSTR + kk] = p;
      p[0] = (bf16)f0.w; p[1] = (bf16)f1.w; *(v2bf*)&Bt[(n + 3) * BSTR + kk] = p;
    }
    __syncthreads();

    v16bf af[2];
    af[0] = gather16(&As[(wv32 * 32 + l15) * ASTR], o0, o1);
    af[1] = gather16(&As[(wv32 * 32 + 16 + l15) * ASTR], o0, o1);
    v16bf bfr[4];
#pragma unroll
    for (int nf = 0; nf < 4; ++nf)
      bfr[nf] = gather16(&Bt[(nf * 16 + l15) * BSTR], o0, o1);
#pragma unroll
    for (int ms = 0; ms < 2; ++ms)
#pragma unroll
      for (int nf = 0; nf < 4; ++nf)
        acc[ms][nf] = WMMA_BF16(af[ms], bfr[nf], acc[ms][nf]);
  }

#pragma unroll
  for (int nf = 0; nf < 4; ++nf) {
    float bi = bia[colBase + nf * 16 + l15];
#pragma unroll
    for (int ms = 0; ms < 2; ++ms)
#pragma unroll
      for (int i = 0; i < 8; ++i) {
        int m = ms * 16 + i + lh * 8;
        out[(size_t)(rowBase + wv32 * 32 + m) * E + colBase + nf * 16 + l15] =
            acc[ms][nf][i] + bi;
      }
  }
}

// ---------------------------------------------------------------------------
extern "C" void kernel_launch(void* const* d_in, const int* in_sizes, int n_in,
                              void* d_out, int out_size, void* d_ws, size_t ws_size,
                              hipStream_t stream) {
  const float* hs   = (const float*)d_in[0];
  const float* rope = (const float*)d_in[1];
  const float* wq   = (const float*)d_in[2];
  const float* bq   = (const float*)d_in[3];
  const float* wk   = (const float*)d_in[4];
  const float* bk   = (const float*)d_in[5];
  const float* wvp  = (const float*)d_in[6];
  const float* bv   = (const float*)d_in[7];
  const float* wo   = (const float*)d_in[8];
  const float* bo   = (const float*)d_in[9];

  const size_t mat = (size_t)BS * E;                 // 8M elems per bf16 matrix
  bf16* wsQ = (bf16*)d_ws;
  bf16* wsK = wsQ + mat;
  bf16* wsV = wsK + mat;
  bf16* wsA = wsV + mat;

  qkv_rope_kernel<<<dim3(BS / 128, H, 3), 128, 0, stream>>>(
      hs, rope, wq, bq, wk, bk, wvp, bv, wsQ, wsK, wsV);
  flash_attn_kernel<<<dim3(S / 128, B * H), 256, 0, stream>>>(wsQ, wsK, wsV, wsA);
  out_proj_kernel<<<dim3(BS / 128, E / 64), 128, 0, stream>>>(wsA, wo, bo, (float*)d_out);
}
